// BiDACPI_69011534512272
// MI455X (gfx1250) — matrix operand order repaired and checked by reading, not compiled, not run on hardware
//
#include <hip/hip_runtime.h>
#include <hip/hip_bf16.h>

// ---------------------------------------------------------------------------
// BiDACPI forward for MI455X (gfx1250, wave32, WMMA).
// - All GEMMs: v_wmma_f32_16x16x32_f16, 4 waves/block on a 32x32 C tile,
//   vectorized b128 global loads -> f16 LDS tiles.
// - Bidirectional attention fused flash-style; tile staging uses gfx1250
//   GLOBAL_LOAD_ASYNC_TO_LDS_B128 with ASYNCcnt-tracked double buffering.
// ---------------------------------------------------------------------------

typedef __attribute__((ext_vector_type(16))) _Float16 v16h;
typedef __attribute__((ext_vector_type(8)))  _Float16 h8;
typedef __attribute__((ext_vector_type(8)))  float    v8f;

#define CB     64
#define CN     256
#define CL     2000
#define CCOMP  64
#define CPROT  128
#define CGATD  64
#define CHEADS 4
#define CLAT   64
#define CNBI   4

__device__ __forceinline__ v8f wmma16(v16h a, v16h b, v8f c) {
  return __builtin_amdgcn_wmma_f32_16x16x32_f16(false, a, false, b, (short)0, c,
                                                false, false);
}

// A fragment 16x32 f16 from row-major source (ld halfs). k = 16*(lane/16) + s;
// per lane one contiguous 32B run -> two b128 loads.
__device__ __forceinline__ v16h load_A_frag(const _Float16* a, int ld) {
  int lane = threadIdx.x & 31;
  int row = lane & 15;
  int kb = (lane >> 4) << 4;
  const _Float16* p = a + row * ld + kb;
  h8 lo = *(const h8*)p;
  h8 hi = *(const h8*)(p + 8);
  v16h r;
#pragma unroll
  for (int i = 0; i < 8; ++i) { r[i] = lo[i]; r[i + 8] = hi[i]; }
  return r;
}

// B fragment 32x16 f16, row-major B[k][n] (ld halfs), unguarded.
__device__ __forceinline__ v16h load_B_frag(const _Float16* b, int ld) {
  int lane = threadIdx.x & 31;
  int col = lane & 15;
  int kb = (lane >> 4) << 4;
  v16h r;
#pragma unroll
  for (int v = 0; v < 8; ++v) {
    r[2 * v]     = b[(kb + 2 * v) * ld + col];
    r[2 * v + 1] = b[(kb + 2 * v + 1) * ld + col];
  }
  return r;
}

// Transposed-B fragment: B[k][n] = src[n*64 + k]; one 32B run per lane.
__device__ __forceinline__ v16h load_BT16(const _Float16* src) {
  int lane = threadIdx.x & 31;
  return *(const v16h*)(src + (long long)(lane & 15) * 64 + ((lane >> 4) << 4));
}

__device__ __forceinline__ void cvt_store8(_Float16* d, float4 a, float4 b) {
  h8 h;
  h[0] = (_Float16)a.x; h[1] = (_Float16)a.y; h[2] = (_Float16)a.z; h[3] = (_Float16)a.w;
  h[4] = (_Float16)b.x; h[5] = (_Float16)b.y; h[6] = (_Float16)b.z; h[7] = (_Float16)b.w;
  *(h8*)d = h;
}

// --- gfx1250 async global->LDS (ISA 15.18.3 op 98; tracked by ASYNCcnt) ----
__device__ __forceinline__ void async_ld16(_Float16* ldsdst, const _Float16* gsrc) {
  unsigned la = (unsigned)(unsigned long long)(void*)ldsdst;  // low 32 = LDS addr
  asm volatile("global_load_async_to_lds_b128 %0, %1, off"
               :: "v"(la), "v"((unsigned long long)(const void*)gsrc)
               : "memory");
}
__device__ __forceinline__ void wait_async0() {
  asm volatile("s_wait_asynccnt 0x0" ::: "memory");
}
__device__ __forceinline__ void wait_async4() {
  asm volatile("s_wait_asynccnt 0x4" ::: "memory");
}

// ---------------------------------------------------------------------------
// Strided-batched GEMM: C = act(A(MxK) @ B(KxNc) + bias).
// 128 threads / 4 waves; block computes 32x32 C; wave w -> 16x16 subtile.
// Requires M%32==0, Nc%32==0, K%32==0 (all call sites comply).
// act: 0 none, 1 leaky(alpha), 2 tanh, 3 elu.
// ---------------------------------------------------------------------------
__global__ void gemm_wmma(const float* __restrict__ A, const float* __restrict__ Bm,
                          void* __restrict__ Cv, const float* __restrict__ bias,
                          int M, int Nc, int K, int lda, int ldb, int ldc,
                          long long sA, long long sB, long long sC,
                          int act, float alpha, int out_half) {
  __shared__ _Float16 As[32 * 32];
  __shared__ _Float16 Bs[32 * 32];
  int bz = blockIdx.z;
  const float* Ab = A + (long long)bz * sA + (long long)blockIdx.x * 32 * lda;
  const float* Bb = Bm + (long long)bz * sB + blockIdx.y * 32;
  int tid = threadIdx.x, lane = tid & 31, wave = tid >> 5;
  int wm = (wave >> 1) << 4, wn = (wave & 1) << 4;
  int lr = tid >> 2;             // 0..31: row for cooperative loads
  int lc = (tid & 3) << 3;       // 0,8,16,24
  v8f acc = {};
  for (int k0 = 0; k0 < K; k0 += 32) {
    const float* sa = Ab + (long long)lr * lda + k0 + lc;
    float4 a0 = *(const float4*)sa;
    float4 a1 = *(const float4*)(sa + 4);
    const float* sb = Bb + (long long)(k0 + lr) * ldb + lc;
    float4 b0 = *(const float4*)sb;
    float4 b1 = *(const float4*)(sb + 4);
    cvt_store8(As + lr * 32 + lc, a0, a1);
    cvt_store8(Bs + lr * 32 + lc, b0, b1);
    __syncthreads();
    v16h af = load_A_frag(As + wm * 32, 32);
    v16h bf = load_B_frag(Bs + wn, 32);
    acc = wmma16(af, bf, acc);
    __syncthreads();
  }
  int n = lane & 15, mb = (lane >> 4) << 3;
  long long gm0 = (long long)blockIdx.x * 32 + wm;
  int gn = blockIdx.y * 32 + wn + n;
  float bval = (bias != nullptr) ? bias[gn] : 0.f;
#pragma unroll
  for (int v = 0; v < 8; ++v) {
    float x = acc[v] + bval;
    if (act == 1)      x = x > 0.f ? x : alpha * x;
    else if (act == 2) x = tanhf(x);
    else if (act == 3) x = x > 0.f ? x : (expf(x) - 1.f);
    long long off = (long long)bz * sC + (gm0 + mb + v) * ldc + gn;
    if (out_half) ((_Float16*)Cv)[off] = (_Float16)x;
    else          ((float*)Cv)[off] = x;
  }
}

// ---------------------------------------------------------------------------
__global__ void gather_rows(const int* __restrict__ idx, const float* __restrict__ emb,
                            float* __restrict__ out, long long rows, int F) {
  long long t = (long long)blockIdx.x * blockDim.x + threadIdx.x;
  if (t >= rows * (long long)F) return;
  long long r = t / F; int f = (int)(t % F);
  out[t] = emb[(long long)idx[r] * F + f];
}

__global__ void evec_kernel(const float* __restrict__ Wh, const float* __restrict__ a,
                            float* __restrict__ e1, float* __restrict__ e2,
                            long long rows) {
  long long r = (long long)blockIdx.x * blockDim.x + threadIdx.x;
  if (r >= rows) return;
  const float* w = Wh + r * 64;
  float s1 = 0.f, s2 = 0.f;
#pragma unroll 8
  for (int d = 0; d < 64; ++d) { s1 += w[d] * a[d]; s2 += w[d] * a[64 + d]; }
  e1[r] = s1; e2[r] = s2;
}

// GAT attention row: softmax_m( adj ? leaky(e1[n]+e2[m]) : -9e15 )
__global__ void att_kernel(const float* __restrict__ e1, const float* __restrict__ e2,
                           const int* __restrict__ adj, float* __restrict__ att) {
  __shared__ float red[256];
  int n = blockIdx.x, b = blockIdx.y, m = threadIdx.x;
  long long row = (long long)b * CN + n;
  float v = e1[row] + e2[(long long)b * CN + m];
  v = v > 0.f ? v : 0.2f * v;
  float x = (adj[row * CN + m] > 0) ? v : -9e15f;
  red[m] = x; __syncthreads();
  for (int s = 128; s > 0; s >>= 1) { if (m < s) red[m] = fmaxf(red[m], red[m + s]); __syncthreads(); }
  float mx = red[0]; __syncthreads();
  float ex = expf(x - mx);
  red[m] = ex; __syncthreads();
  for (int s = 128; s > 0; s >>= 1) { if (m < s) red[m] += red[m + s]; __syncthreads(); }
  att[row * CN + m] = ex / red[0];
}

// 11x11 single-channel conv over (L, PROT) + bias + leaky(0.2)
__global__ void conv11_kernel(const float* __restrict__ in, float* __restrict__ out,
                              const float* __restrict__ w, const float* __restrict__ bias,
                              int bi) {
  __shared__ float ws[121];
  if (threadIdx.x < 121) ws[threadIdx.x] = w[threadIdx.x];
  __syncthreads();
  long long t = (long long)blockIdx.x * blockDim.x + threadIdx.x;
  long long total = (long long)CB * CL * CPROT;
  if (t >= total) return;
  int p = (int)(t % CPROT);
  long long rem = t / CPROT;
  int l = (int)(rem % CL);
  int b = (int)(rem / CL);
  float acc = bias[bi];
  const float* inb = in + (long long)b * CL * CPROT;
  for (int dy = -5; dy <= 5; ++dy) {
    int ll = l + dy; if (ll < 0 || ll >= CL) continue;
    const float* rowp = inb + (long long)ll * CPROT;
    const float* wrow = ws + (dy + 5) * 11;
#pragma unroll
    for (int dx = -5; dx <= 5; ++dx) {
      int pp = p + dx;
      if (pp >= 0 && pp < CPROT) acc += rowp[pp] * wrow[dx + 5];
    }
  }
  out[t] = acc > 0.f ? acc : 0.2f * acc;
}

__global__ void f2h_kernel(const float* __restrict__ in, _Float16* __restrict__ out,
                           long long n) {
  long long t = (long long)blockIdx.x * blockDim.x + threadIdx.x;
  if (t < n) out[t] = (_Float16)in[t];
}

// ---------------------------------------------------------------------------
// ctrans: atoms_trans[b,n,:] = sum_l tanh(au[n].amino[l]) * mask * tp[l,:]
// 128 threads / 4 waves; amino & tp tiles (32x64) double-buffered via async
// LDS loads (4 async ops per wave per stage; loads complete in order, so
// s_wait_asynccnt 4 releases the older stage while the next stays in flight).
// Rows past L are clamped (finite data); pmask zeroes those A columns, so
// out-of-range contributions are exactly 0.
// ---------------------------------------------------------------------------
__global__ void ctrans_kernel(const _Float16* __restrict__ au,
                              const _Float16* __restrict__ amh,
                              const _Float16* __restrict__ tph,
                              const float* __restrict__ amask,
                              const float* __restrict__ pmask,
                              float* __restrict__ outT) {
  __shared__ _Float16 Ams[2][32 * 64];
  __shared__ _Float16 Ts[2][32 * 64];
  __shared__ _Float16 Af[4 * 16 * 32];
  int b = blockIdx.y;
  int tid = threadIdx.x, lane = tid & 31, wave = tid >> 5;
  int n0 = blockIdx.x * 64 + wave * 16;
  const _Float16* aub = au + ((long long)b * CN + n0) * 64;
  const _Float16* amb = amh + (long long)b * CL * 64;
  const _Float16* tpb = tph + (long long)b * CL * 64;
  // staging geometry: threads 0..63 -> amino tile, 64..127 -> tp tile
  int sg = tid & 63, sr = sg >> 1, sch = (sg & 1) << 5;
  bool samino = tid < 64;
  const _Float16* sbase = samino ? amb : tpb;

  v16h a0 = load_A_frag(aub, 64);
  v16h a1 = load_A_frag(aub + 32, 64);
  int nloc = lane & 15, mb = (lane >> 4) << 3;
  _Float16* Afw = Af + wave * 512;
  float rowm[8];
#pragma unroll
  for (int v = 0; v < 8; ++v) rowm[v] = amask[(long long)b * CN + n0 + mb + v];
  v8f acc0 = {}, acc1 = {}, acc2 = {}, acc3 = {};

  const int NIT = (CL + 31) / 32;   // 63
  // prologue: stage tiles 0 and 1
#pragma unroll
  for (int s = 0; s < 2; ++s) {
    int l = s * 32 + sr;
    const _Float16* src = sbase + (long long)l * 64 + sch;
    _Float16* dst = (samino ? Ams[s] : Ts[s]) + sr * 64 + sch;
#pragma unroll
    for (int c = 0; c < 4; ++c) async_ld16(dst + c * 8, src + c * 8);
  }
  for (int it = 0; it < NIT; ++it) {
    int l0 = it * 32, cur = it & 1;
    if (it + 1 < NIT) wait_async4(); else wait_async0();
    __syncthreads();
    const _Float16* AmsC = Ams[cur];
    const _Float16* TsC  = Ts[cur];
    v8f Al = {}, Ar = {};
    Al = wmma16(a0, load_BT16(AmsC), Al);
    Al = wmma16(a1, load_BT16(AmsC + 32), Al);
    Ar = wmma16(a0, load_BT16(AmsC + 16 * 64), Ar);
    Ar = wmma16(a1, load_BT16(AmsC + 16 * 64 + 32), Ar);
#pragma unroll
    for (int v = 0; v < 8; ++v) {
      int lL = l0 + nloc, lR = l0 + 16 + nloc;
      float mL = (lL < CL) ? pmask[(long long)b * CL + lL] : 0.f;
      float mR = (lR < CL) ? pmask[(long long)b * CL + lR] : 0.f;
      Afw[(mb + v) * 32 + nloc]      = (_Float16)(tanhf(Al[v]) * rowm[v] * mL);
      Afw[(mb + v) * 32 + 16 + nloc] = (_Float16)(tanhf(Ar[v]) * rowm[v] * mR);
    }
    __syncthreads();
    v16h af = load_A_frag(Afw, 32);
    acc0 = wmma16(af, load_B_frag(TsC + 0,  64), acc0);
    acc1 = wmma16(af, load_B_frag(TsC + 16, 64), acc1);
    acc2 = wmma16(af, load_B_frag(TsC + 32, 64), acc2);
    acc3 = wmma16(af, load_B_frag(TsC + 48, 64), acc3);
    __syncthreads();
    if (it + 2 < NIT) {   // prefetch tile it+2 into the slot just released
      int l = l0 + 64 + sr; if (l >= CL) l = CL - 1;
      const _Float16* src = sbase + (long long)l * 64 + sch;
      _Float16* dst = (samino ? Ams[cur] : Ts[cur]) + sr * 64 + sch;
#pragma unroll
      for (int c = 0; c < 4; ++c) async_ld16(dst + c * 8, src + c * 8);
    }
  }
  float* ob = outT + ((long long)b * CN + n0) * 64;
#pragma unroll
  for (int v = 0; v < 8; ++v) {
    ob[(mb + v) * 64 + nloc]      = acc0[v];
    ob[(mb + v) * 64 + 16 + nloc] = acc1[v];
    ob[(mb + v) * 64 + 32 + nloc] = acc2[v];
    ob[(mb + v) * 64 + 48 + nloc] = acc3[v];
  }
}

// ---------------------------------------------------------------------------
// ptrans: amino_trans[b,l,:] = sum_n tanh(amino[l].au[n]) * mask * tc[n,:]
// 160 threads / 5 waves (125 tiles = 25 blocks x 5 waves, all valid).
// au & tc tiles double-buffered via async LDS loads (wave 4 stages nothing;
// its ASYNCcnt waits are trivially satisfied).
// ---------------------------------------------------------------------------
__global__ void ptrans_kernel(const _Float16* __restrict__ amh,
                              const _Float16* __restrict__ au,
                              const _Float16* __restrict__ tch,
                              const float* __restrict__ pmask,
                              const float* __restrict__ amask,
                              float* __restrict__ outT) {
  __shared__ _Float16 Aus[2][32 * 64];
  __shared__ _Float16 Ts[2][32 * 64];
  __shared__ _Float16 Af[5 * 16 * 32];
  int b = blockIdx.y;
  int tid = threadIdx.x, lane = tid & 31, wave = tid >> 5;
  int l0 = (blockIdx.x * 5 + wave) * 16;
  const _Float16* amb = amh + ((long long)b * CL + l0) * 64;
  const _Float16* aub = au + (long long)b * CN * 64;
  const _Float16* tcb = tch + (long long)b * CN * 64;
  int sg = tid & 63, sr = sg >> 1, sch = (sg & 1) << 5;
  bool sau = tid < 64;
  const _Float16* sbase = sau ? aub : tcb;

  v16h a0 = load_A_frag(amb, 64);
  v16h a1 = load_A_frag(amb + 32, 64);
  int nloc = lane & 15, mb = (lane >> 4) << 3;
  _Float16* Afw = Af + wave * 512;
  float rowm[8];
#pragma unroll
  for (int v = 0; v < 8; ++v) rowm[v] = pmask[(long long)b * CL + l0 + mb + v];
  v8f acc0 = {}, acc1 = {}, acc2 = {}, acc3 = {};

  const int NIT = CN / 32;   // 8
#pragma unroll
  for (int s = 0; s < 2; ++s) {
    if (tid < 128) {
      const _Float16* src = sbase + (long long)(s * 32 + sr) * 64 + sch;
      _Float16* dst = (sau ? Aus[s] : Ts[s]) + sr * 64 + sch;
#pragma unroll
      for (int c = 0; c < 4; ++c) async_ld16(dst + c * 8, src + c * 8);
    }
  }
  for (int it = 0; it < NIT; ++it) {
    int n0 = it * 32, cur = it & 1;
    if (it + 1 < NIT) wait_async4(); else wait_async0();
    __syncthreads();
    const _Float16* AusC = Aus[cur];
    const _Float16* TsC  = Ts[cur];
    v8f Al = {}, Ar = {};
    Al = wmma16(a0, load_BT16(AusC), Al);
    Al = wmma16(a1, load_BT16(AusC + 32), Al);
    Ar = wmma16(a0, load_BT16(AusC + 16 * 64), Ar);
    Ar = wmma16(a1, load_BT16(AusC + 16 * 64 + 32), Ar);
#pragma unroll
    for (int v = 0; v < 8; ++v) {
      float mL = amask[(long long)b * CN + n0 + nloc];
      float mR = amask[(long long)b * CN + n0 + 16 + nloc];
      Afw[(mb + v) * 32 + nloc]      = (_Float16)(tanhf(Al[v]) * rowm[v] * mL);
      Afw[(mb + v) * 32 + 16 + nloc] = (_Float16)(tanhf(Ar[v]) * rowm[v] * mR);
    }
    __syncthreads();
    v16h af = load_A_frag(Afw, 32);
    acc0 = wmma16(af, load_B_frag(TsC + 0,  64), acc0);
    acc1 = wmma16(af, load_B_frag(TsC + 16, 64), acc1);
    acc2 = wmma16(af, load_B_frag(TsC + 32, 64), acc2);
    acc3 = wmma16(af, load_B_frag(TsC + 48, 64), acc3);
    __syncthreads();
    if (it + 2 < NIT && tid < 128) {
      const _Float16* src = sbase + (long long)(n0 + 64 + sr) * 64 + sch;
      _Float16* dst = (sau ? Aus[cur] : Ts[cur]) + sr * 64 + sch;
#pragma unroll
      for (int c = 0; c < 4; ++c) async_ld16(dst + c * 8, src + c * 8);
    }
  }
  float* ob = outT + ((long long)b * CL + l0) * 64;
#pragma unroll
  for (int v = 0; v < 8; ++v) {
    ob[(mb + v) * 64 + nloc]      = acc0[v];
    ob[(mb + v) * 64 + 16 + nloc] = acc1[v];
    ob[(mb + v) * 64 + 32 + nloc] = acc2[v];
    ob[(mb + v) * 64 + 48 + nloc] = acc3[v];
  }
}

// score[r] = q[r,:].w[0:64] + tr[r,:].w[64:128] + b
__global__ void score_kernel(const float* __restrict__ q, const float* __restrict__ tr,
                             const float* __restrict__ w, const float* __restrict__ bb,
                             float* __restrict__ s, long long rows) {
  long long r = (long long)blockIdx.x * blockDim.x + threadIdx.x;
  if (r >= rows) return;
  float acc = bb[0];
#pragma unroll 8
  for (int d = 0; d < 64; ++d)
    acc += q[r * 64 + d] * w[d] + tr[r * 64 + d] * w[64 + d];
  s[r] = acc;
}

// masked softmax over R + weighted sum of vec rows -> out[b, col0:col0+64]
__global__ void attpool_kernel(const float* __restrict__ s, const float* __restrict__ mask,
                               const float* __restrict__ vec, float* __restrict__ out,
                               int R, int ldo, int col0) {
  __shared__ float wts[2048];
  __shared__ float red[256];
  int b = blockIdx.x, t = threadIdx.x;
  const float* sb = s + (long long)b * R;
  const float* mb = mask + (long long)b * R;
  float mx = -3.4e38f;
  for (int r = t; r < R; r += 256) mx = fmaxf(mx, sb[r]);
  red[t] = mx; __syncthreads();
  for (int st = 128; st > 0; st >>= 1) { if (t < st) red[t] = fmaxf(red[t], red[t + st]); __syncthreads(); }
  mx = red[0]; __syncthreads();
  float sum = 0.f;
  for (int r = t; r < R; r += 256) { float ex = expf(sb[r] - mx) * mb[r]; wts[r] = ex; sum += ex; }
  red[t] = sum; __syncthreads();
  for (int st = 128; st > 0; st >>= 1) { if (t < st) red[t] += red[t + st]; __syncthreads(); }
  float denom = red[0] + 1e-6f;
  __syncthreads();
  if (t < 64) {
    float acc = 0.f;
    const float* vb = vec + (long long)b * R * 64;
    for (int r = 0; r < R; ++r) acc += vb[(long long)r * 64 + t] * wts[r];
    out[(long long)b * ldo + col0 + t] = acc / denom;
  }
}

// out[b] = sum_{p,q} leaky(cf[b,p]*pf[b,q], 0.1) * w[p*64+q] + bias
__global__ void final_kernel(const float* __restrict__ cf, const float* __restrict__ pf,
                             const float* __restrict__ w, const float* __restrict__ bb,
                             float* __restrict__ out) {
  __shared__ float red[256];
  int b = blockIdx.x, t = threadIdx.x;
  float acc = 0.f;
  for (int idx = t; idx < 128 * 64; idx += 256) {
    int p = idx >> 6, q = idx & 63;
    float x = cf[b * 128 + p] * pf[b * 64 + q];
    x = x > 0.f ? x : 0.1f * x;
    acc += x * w[idx];
  }
  red[t] = acc; __syncthreads();
  for (int s = 128; s > 0; s >>= 1) { if (t < s) red[t] += red[t + s]; __syncthreads(); }
  if (t == 0) out[b] = red[0] + bb[0];
}

// ---------------------------------------------------------------------------

extern "C" void kernel_launch(void* const* d_in, const int* in_sizes, int n_in,
                              void* d_out, int out_size, void* d_ws, size_t ws_size,
                              hipStream_t stream) {
  const int*   atoms      = (const int*)d_in[0];
  const float* atoms_mask = (const float*)d_in[1];
  const int*   adjacency  = (const int*)d_in[2];
  const int*   amino      = (const int*)d_in[3];
  const float* amino_mask = (const float*)d_in[4];
  const float* fps        = (const float*)d_in[5];
  const float* emb_atom   = (const float*)d_in[6];
  const float* emb_amino  = (const float*)d_in[7];
  const float* gat_W      = (const float*)d_in[8];
  const float* gat_a      = (const float*)d_in[9];
  const float* gout_W     = (const float*)d_in[10];
  const float* gout_a     = (const float*)d_in[11];
  const float* Wcomp_w    = (const float*)d_in[12];
  const float* Wcomp_b    = (const float*)d_in[13];
  const float* conv_w     = (const float*)d_in[14];
  const float* conv_b     = (const float*)d_in[15];
  const float* Wprot_w    = (const float*)d_in[16];
  const float* Wprot_b    = (const float*)d_in[17];
  const float* fp0        = (const float*)d_in[18];
  const float* fp1        = (const float*)d_in[19];
  const float* U          = (const float*)d_in[20];
  const float* tp2c_w     = (const float*)d_in[21];
  const float* tp2c_b     = (const float*)d_in[22];
  const float* tc2p_w     = (const float*)d_in[23];
  const float* tc2p_b     = (const float*)d_in[24];
  const float* bhc_w      = (const float*)d_in[25];
  const float* bhc_b      = (const float*)d_in[26];
  const float* bhp_w      = (const float*)d_in[27];
  const float* bhp_b      = (const float*)d_in[28];
  const float* battc_w    = (const float*)d_in[29];
  const float* battc_b    = (const float*)d_in[30];
  const float* battp_w    = (const float*)d_in[31];
  const float* battp_b    = (const float*)d_in[32];
  const float* combc_w    = (const float*)d_in[33];
  const float* combc_b    = (const float*)d_in[34];
  const float* combp_w    = (const float*)d_in[35];
  const float* combp_b    = (const float*)d_in[36];
  const float* out_w      = (const float*)d_in[37];
  const float* out_b      = (const float*)d_in[38];
  float* out = (float*)d_out;

  const long long BN = (long long)CB * CN;   // 16384
  const long long BL = (long long)CB * CL;   // 128000

  float* ws = (float*)d_ws;
  long long off = 0;
  auto alloc = [&](long long n) { float* p = ws + off; off += n; return p; };

  float* buf0   = alloc(BL * CPROT);         // CNN ping (reused as f16 staging)
  float* buf1   = alloc(BL * CPROT);         // CNN pong
  float* avb    = alloc(BN * CCOMP);         // atom embeddings / hp2 temp
  float* WhB    = alloc((long long)CHEADS * BN * CGATD);
  float* attB   = alloc(BN * CN);
  float* multiB = alloc(BN * CHEADS * CGATD);
  float* atomsV = alloc(BN * CLAT);
  float* aminoV = alloc(BL * CLAT);
  float* aminoT = alloc(BL * CLAT);          // amino_trans
  float* atomsT = alloc(BN * CLAT);          // atoms_trans
  float* e1 = alloc(BN);
  float* e2 = alloc(BN);
  float* sc = alloc(BN);
  float* sp = alloc(BL);
  float* catC = alloc((long long)CB * CLAT * CNBI);
  float* catP = alloc((long long)CB * CLAT * CNBI);
  float* cfF  = alloc((long long)CB * 2 * CLAT);
  float* pfF  = alloc((long long)CB * CLAT);
  float* sfT  = alloc((long long)CB * CLAT);

  // f16 staging aliases inside buf0/buf1 (free after the CNN phase)
  _Float16* aminoH = (_Float16*)buf0;                   // BL*64 halves
  _Float16* tpH    = (_Float16*)(buf0 + BL * CLAT / 2); // BL*64 halves
  float*    pq     = buf0 + BL * CLAT;                  // BL*64 floats
  _Float16* auH    = (_Float16*)buf1;                   // BN*64 halves
  _Float16* tcH    = (_Float16*)(buf1 + BN * CLAT / 2); // BN*64 halves
  float*    bq     = buf1 + BN * CLAT;                  // BN*64 floats

  dim3 blkG(128, 1, 1);

  // ---- comp GAT ----
  gather_rows<<<(unsigned)((BN * CCOMP + 255) / 256), 256, 0, stream>>>(
      atoms, emb_atom, avb, BN, CCOMP);
  gemm_wmma<<<dim3(512, 2, CHEADS), blkG, 0, stream>>>(
      avb, gat_W, WhB, nullptr, (int)BN, 64, 64, 64, 64, 64,
      0LL, 64LL * 64, BN * 64, 0, 0.f, 0);
  for (int h = 0; h < CHEADS; ++h) {
    evec_kernel<<<64, 256, 0, stream>>>(WhB + (long long)h * BN * 64,
                                        gat_a + h * 128, e1, e2, BN);
    att_kernel<<<dim3(CN, CB), 256, 0, stream>>>(e1, e2, adjacency, attB);
    gemm_wmma<<<dim3(8, 2, CB), blkG, 0, stream>>>(
        attB, WhB + (long long)h * BN * 64, multiB + h * 64, nullptr,
        CN, 64, CN, CN, 64, CHEADS * CGATD,
        (long long)CN * CN, (long long)CN * 64, (long long)CN * CHEADS * CGATD,
        3, 0.f, 0);
  }
  gemm_wmma<<<dim3(512, 2, 1), blkG, 0, stream>>>(
      multiB, gout_W, WhB, nullptr, (int)BN, 64, 256, 256, 64, 64,
      0LL, 0LL, 0LL, 0, 0.f, 0);
  evec_kernel<<<64, 256, 0, stream>>>(WhB, gout_a, e1, e2, BN);
  att_kernel<<<dim3(CN, CB), 256, 0, stream>>>(e1, e2, adjacency, attB);
  gemm_wmma<<<dim3(8, 2, CB), blkG, 0, stream>>>(
      attB, WhB, avb, nullptr, CN, 64, CN, CN, 64, 64,
      (long long)CN * CN, (long long)CN * 64, (long long)CN * 64, 3, 0.f, 0);
  gemm_wmma<<<dim3(512, 2, 1), blkG, 0, stream>>>(
      avb, Wcomp_w, atomsV, Wcomp_b, (int)BN, 64, 64, 64, 64, 64,
      0LL, 0LL, 0LL, 1, 0.2f, 0);

  // ---- prot CNN ----
  gather_rows<<<(unsigned)((BL * CPROT + 255) / 256), 256, 0, stream>>>(
      amino, emb_amino, buf0, BL, CPROT);
  conv11_kernel<<<64000, 256, 0, stream>>>(buf0, buf1, conv_w, conv_b, 0);
  conv11_kernel<<<64000, 256, 0, stream>>>(buf1, buf0, conv_w + 121, conv_b, 1);
  conv11_kernel<<<64000, 256, 0, stream>>>(buf0, buf1, conv_w + 242, conv_b, 2);
  gemm_wmma<<<dim3(4000, 2, 1), blkG, 0, stream>>>(
      buf1, Wprot_w, aminoV, Wprot_b, (int)BL, 64, 128, 128, 64, 64,
      0LL, 0LL, 0LL, 1, 0.2f, 0);

  // ---- fingerprint MLP -> cf_final[:, 64:128] ----
  gemm_wmma<<<dim3(2, 2, 1), blkG, 0, stream>>>(
      fps, fp0, sfT, nullptr, CB, 64, 1024, 1024, 64, 64, 0LL, 0LL, 0LL, 1, 0.1f, 0);
  gemm_wmma<<<dim3(2, 2, 1), blkG, 0, stream>>>(
      sfT, fp1, (void*)(cfF + 64), nullptr, CB, 64, 64, 64, 64, 128,
      0LL, 0LL, 0LL, 1, 0.1f, 0);

  // ---- bidirectional attention ----
  f2h_kernel<<<(unsigned)((BL * 64 + 255) / 256), 256, 0, stream>>>(
      aminoV, aminoH, BL * 64);
  for (int i = 0; i < CNBI; ++i) {
    gemm_wmma<<<dim3(512, 2, 1), blkG, 0, stream>>>(
        atomsV, U + i * 4096, auH, nullptr, (int)BN, 64, 64, 64, 64, 64,
        0LL, 0LL, 0LL, 0, 0.f, 1);
    gemm_wmma<<<dim3(4000, 2, 1), blkG, 0, stream>>>(
        aminoV, tp2c_w + i * 4096, tpH, tp2c_b + i * 64, (int)BL, 64, 64, 64, 64, 64,
        0LL, 0LL, 0LL, 2, 0.f, 1);
    gemm_wmma<<<dim3(512, 2, 1), blkG, 0, stream>>>(
        atomsV, tc2p_w + i * 4096, tcH, tc2p_b + i * 64, (int)BN, 64, 64, 64, 64, 64,
        0LL, 0LL, 0LL, 2, 0.f, 1);
    ctrans_kernel<<<dim3(CN / 64, CB), 128, 0, stream>>>(
        auH, aminoH, tpH, atoms_mask, amino_mask, atomsT);
    ptrans_kernel<<<dim3(25, CB), 160, 0, stream>>>(
        aminoH, auH, tcH, amino_mask, atoms_mask, aminoT);
    gemm_wmma<<<dim3(512, 2, 1), blkG, 0, stream>>>(
        atomsV, bhc_w + i * 4096, bq, bhc_b + i * 64, (int)BN, 64, 64, 64, 64, 64,
        0LL, 0LL, 0LL, 2, 0.f, 0);
    gemm_wmma<<<dim3(4000, 2, 1), blkG, 0, stream>>>(
        aminoV, bhp_w + i * 4096, pq, bhp_b + i * 64, (int)BL, 64, 64, 64, 64, 64,
        0LL, 0LL, 0LL, 2, 0.f, 0);
    score_kernel<<<64, 256, 0, stream>>>(bq, atomsT, battc_w + i * 128,
                                         battc_b + i, sc, BN);
    score_kernel<<<500, 256, 0, stream>>>(pq, aminoT, battp_w + i * 128,
                                          battp_b + i, sp, BL);
    attpool_kernel<<<CB, 256, 0, stream>>>(sc, atoms_mask, atomsV, catC,
                                           CN, CNBI * CLAT, i * CLAT);
    attpool_kernel<<<CB, 256, 0, stream>>>(sp, amino_mask, aminoV, catP,
                                           CL, CNBI * CLAT, i * CLAT);
  }

  // ---- combine + output ----
  gemm_wmma<<<dim3(2, 2, 1), blkG, 0, stream>>>(
      catC, combc_w, cfF, combc_b, CB, 64, 256, 256, 64, 128, 0LL, 0LL, 0LL, 0, 0.f, 0);
  gemm_wmma<<<dim3(2, 2, 1), blkG, 0, stream>>>(
      catP, combp_w, pfF, combp_b, CB, 64, 256, 256, 64, 64, 0LL, 0LL, 0LL, 0, 0.f, 0);
  final_kernel<<<CB, 256, 0, stream>>>(cfF, pfF, out_w, out_b, out);
}
